// SimpleQuantize_7155415515597
// MI455X (gfx1250) — matrix operand order repaired and checked by reading, not compiled, not run on hardware
//
#include <hip/hip_runtime.h>
#include <hip/hip_bf16.h>

// ---------------------------------------------------------------------------
// VQ quantize for MI455X (gfx1250, wave32, WMMA + async-to-LDS).
//   scores = input[16384,256] @ weight[8192,256]^T   (bf16x3 split precision)
//   idxs   = argmax_v scores                          (fused, running)
//   quant  = weight[idxs]                             (exact f32 gather)
//
// Blocking: 4 waves/block own 4 different 16-row M-strips; the B tile
// (16 vocab rows, hi+lo planes, 16 KB) is staged ONCE per block into LDS with
// GLOBAL_LOAD_ASYNC_TO_LDS_B128 (double-buffered), cutting L2 B-traffic 4x
// (8 GB -> 2 GB) -- the WMMA work itself is only ~12.6M v_wmma instructions.
// ---------------------------------------------------------------------------

typedef __attribute__((ext_vector_type(16))) __bf16 v16bf;
typedef __attribute__((ext_vector_type(8)))  __bf16 v8bf;
typedef __attribute__((ext_vector_type(8)))  float  v8f;
typedef __attribute__((ext_vector_type(4)))  int    v4i;

#define EMBD_DIM   256
#define VOCAB      8192
#define MROWS      16384            // B*S = 8*2048
#define NTILES     (VOCAB / 16)     // 512
#define KSTEPS     (EMBD_DIM / 32)  // 8
#define WAVES_PB   4
#define TILE_BYTES 16384            // B tile in LDS: 16 rows * 256 K * 2 planes * 2B

// ---------------------------------------------------------------------------
// CDNA5 async global->LDS copy (ASYNCcnt path), builtin if present else asm.
// Builtin signature (from hipcc diagnostic): (AS1 v4i*, AS3 v4i*, imm, imm).
// ---------------------------------------------------------------------------
__device__ __forceinline__ void async_copy_b128(const void* gsrc, void* ldst) {
#if __has_builtin(__builtin_amdgcn_global_load_async_to_lds_b128)
  __builtin_amdgcn_global_load_async_to_lds_b128(
      (__attribute__((address_space(1))) v4i*)gsrc,
      (__attribute__((address_space(3))) v4i*)ldst, 0, 0);
#else
  unsigned loff = (unsigned)(size_t)(__attribute__((address_space(3))) void*)ldst;
  unsigned long long ga = (unsigned long long)(size_t)gsrc;
  asm volatile("global_load_async_to_lds_b128 %0, %1, off"
               :: "v"(loff), "v"(ga) : "memory");
#endif
}

__device__ __forceinline__ void wait_async0() {
#if __has_builtin(__builtin_amdgcn_s_wait_asynccnt)
  __builtin_amdgcn_s_wait_asynccnt(0);
#else
  asm volatile("s_wait_asynccnt 0" ::: "memory");
#endif
}

// ---------------------------------------------------------------------------
// Kernel 1: split f32 weight into bf16 hi/lo planes (done once; B is reused
// by every M-strip, so converting in the GEMM loop would be VALU-bound).
// ---------------------------------------------------------------------------
__global__ void vq_split_weight(const float* __restrict__ w,
                                __bf16* __restrict__ whi,
                                __bf16* __restrict__ wlo,
                                int n) {
  int i = blockIdx.x * blockDim.x + threadIdx.x;
  if (i < n) {
    float x  = w[i];
    __bf16 h = (__bf16)x;
    whi[i] = h;
    wlo[i] = (__bf16)(x - (float)h);
  }
}

// ---------------------------------------------------------------------------
// Kernel 2: block of 4 waves; each wave owns a 16-row M strip. Block loops
// over all 512 N-tiles with the B tile double-buffered in LDS in *fragment*
// order [kstep][plane][half][lane]x16B so ds reads are lane-consecutive 16B
// (conflict-free sweeps).
// ---------------------------------------------------------------------------
__global__ void __launch_bounds__(128, 1)
vq_argmax_gather(const float* __restrict__ input,
                 const float* __restrict__ weight,
                 const __bf16* __restrict__ whi,
                 const __bf16* __restrict__ wlo,
                 float* __restrict__ outq,
                 int*   __restrict__ outidx) {
  __shared__ __align__(16) unsigned char sB[2][TILE_BYTES];

  const int lane  = threadIdx.x & 31;
  const int wave  = threadIdx.x >> 5;
  const int strip = blockIdx.x * WAVES_PB + wave;   // 0..1023
  const int m0    = strip * 16;

  const int col   = lane & 15;   // column-within-tile this lane carries (B/C/D)
  const int half  = lane >> 4;   // half-wave id
  const int khalf = half * 8;    // A-fragment K offset for this half-wave

  // ---- Per-thread staging descriptors: 1024 16B chunks / 128 threads = 8.
  //      chunk c: lane_c=c&31, h=(c>>5)&1, q=c>>6 (q = s*2+plane).
  //      LDS offset  = c*16 (fragment order).
  //      Global src  = plane + (v0+col_c)*256 + s*32 + half_c*8 + h*16 (elems).
  const __bf16* gsrc_base[8];
#pragma unroll
  for (int j = 0; j < 8; ++j) {
    const int c      = threadIdx.x + 128 * j;
    const int lane_c = c & 31;
    const int h      = (c >> 5) & 1;
    const int q      = c >> 6;
    const int s      = q >> 1;
    const int pl     = q & 1;
    const __bf16* P  = pl ? wlo : whi;
    gsrc_base[j] = P + (size_t)(lane_c & 15) * EMBD_DIM
                     + s * 32 + (lane_c >> 4) * 8 + h * 16;
  }

  // ---- Load + split the A strip into registers (ISA 16-bit A 16x32 layout:
  //      lane = row m0+col; elems 0..7 -> K=kb+0..7, elems 8..15 -> K=kb+16..23)
  v16bf a_hi[KSTEPS], a_lo[KSTEPS];
  const float* arow = input + (size_t)(m0 + col) * EMBD_DIM;
#pragma unroll
  for (int s = 0; s < KSTEPS; ++s) {
    const int kb = s * 32 + khalf;
#pragma unroll
    for (int e = 0; e < 16; ++e) {
      const int k = kb + (e < 8 ? e : e + 8);
      float  x = arow[k];
      __bf16 h = (__bf16)x;
      a_hi[s][e] = h;
      a_lo[s][e] = (__bf16)(x - (float)h);
    }
  }

  // ---- Running argmax state mirrored on the C layout:
  //      VGPR g, lanes 0-15 -> row m0+g ; lanes 16-31 -> row m0+8+g.
  float best_val[8];
  int   best_idx[8];
#pragma unroll
  for (int g = 0; g < 8; ++g) { best_val[g] = -3.402823466e38f; best_idx[g] = 0; }

  // ---- Prologue: stage tile 0 into buffer 0.
#pragma unroll
  for (int j = 0; j < 8; ++j)
    async_copy_b128(gsrc_base[j], &sB[0][(threadIdx.x + 128 * j) * 16]);
  wait_async0();
  __syncthreads();

  // ---- Main loop over all vocab tiles (weight planes are L2-resident).
  for (int nt = 0; nt < NTILES; ++nt) {
    const int cur = nt & 1;

    // Stage tile nt+1 into the other buffer (overlapped with compute).
    if (nt + 1 < NTILES) {
      const size_t goff = (size_t)(nt + 1) * 16 * EMBD_DIM;  // elems
#pragma unroll
      for (int j = 0; j < 8; ++j)
        async_copy_b128(gsrc_base[j] + goff,
                        &sB[cur ^ 1][(threadIdx.x + 128 * j) * 16]);
    }

    // Compute: fragments from LDS, lane-consecutive 16B chunks.
    const unsigned char* base = sB[cur];
    v8f c = {};
#pragma unroll
    for (int s = 0; s < KSTEPS; ++s) {
      const v8bf h0 = *(const v8bf*)(base + (4 * s + 0) * 512 + lane * 16);
      const v8bf h1 = *(const v8bf*)(base + (4 * s + 1) * 512 + lane * 16);
      const v8bf l0 = *(const v8bf*)(base + (4 * s + 2) * 512 + lane * 16);
      const v8bf l1 = *(const v8bf*)(base + (4 * s + 3) * 512 + lane * 16);
      const v16bf b_hi = __builtin_shufflevector(h0, h1, 0, 1, 2, 3, 4, 5, 6, 7,
                                                 8, 9, 10, 11, 12, 13, 14, 15);
      const v16bf b_lo = __builtin_shufflevector(l0, l1, 0, 1, 2, 3, 4, 5, 6, 7,
                                                 8, 9, 10, 11, 12, 13, 14, 15);
      c = __builtin_amdgcn_wmma_f32_16x16x32_bf16(false, a_hi[s], false, b_hi,
                                                  (short)0, c, false, false);
      c = __builtin_amdgcn_wmma_f32_16x16x32_bf16(false, a_hi[s], false, b_lo,
                                                  (short)0, c, false, false);
      c = __builtin_amdgcn_wmma_f32_16x16x32_bf16(false, a_lo[s], false, b_hi,
                                                  (short)0, c, false, false);
    }

    const int myidx = nt * 16 + col;  // column index this lane carries
#pragma unroll
    for (int g = 0; g < 8; ++g) {
      // strict '>' keeps the earliest (lowest) index within this lane
      if (c[g] > best_val[g]) { best_val[g] = c[g]; best_idx[g] = myidx; }
    }

    wait_async0();     // our chunks of tile nt+1 have landed
    __syncthreads();   // everyone done reading cur + staging visible
  }

  // ---- Reduce argmax across the 16 lanes holding one row's columns.
#pragma unroll
  for (int m = 1; m < 16; m <<= 1) {
#pragma unroll
    for (int g = 0; g < 8; ++g) {
      float ov = __shfl_xor(best_val[g], m, 32);
      int   oi = __shfl_xor(best_idx[g], m, 32);
      if (ov > best_val[g] || (ov == best_val[g] && oi < best_idx[g])) {
        best_val[g] = ov; best_idx[g] = oi;
      }
    }
  }

  // ---- Write indices: lane 0 -> rows m0+0..7, lane 16 -> rows m0+8..15.
  if (col == 0) {
#pragma unroll
    for (int g = 0; g < 8; ++g) outidx[m0 + half * 8 + g] = best_idx[g];
  }

  // ---- Gather winning codebook rows (exact f32), wave-cooperative b128s.
#pragma unroll
  for (int r = 0; r < 16; ++r) {
    const int idx = __shfl(best_idx[r & 7], (r >> 3) * 16, 32);
    const float4* src = (const float4*)(weight + (size_t)idx * EMBD_DIM);
    float4*       dst = (float4*)(outq + (size_t)(m0 + r) * EMBD_DIM);
    dst[lane]      = src[lane];        // bytes   0..511 of the row
    dst[lane + 32] = src[lane + 32];   // bytes 512..1023 of the row
  }
}

// ---------------------------------------------------------------------------
extern "C" void kernel_launch(void* const* d_in, const int* in_sizes, int n_in,
                              void* d_out, int out_size, void* d_ws, size_t ws_size,
                              hipStream_t stream) {
  const float* input  = (const float*)d_in[0];   // [8,2048,256] f32
  const float* weight = (const float*)d_in[1];   // [8192,256]  f32

  // d_out: quantize (16384*256 f32) then idxs (16384 int32), both 4B elems.
  float* outq   = (float*)d_out;
  int*   outidx = (int*)d_out + (size_t)MROWS * EMBD_DIM;

  // Workspace: bf16 hi plane (4MB) + bf16 lo plane (4MB).
  const int nw = VOCAB * EMBD_DIM;               // 2,097,152 elems
  __bf16* whi = (__bf16*)d_ws;
  __bf16* wlo = whi + nw;

  vq_split_weight<<<(nw + 255) / 256, 256, 0, stream>>>(weight, whi, wlo, nw);

  // 1024 strips of 16 rows; 4 waves/block -> 256 blocks x 128 threads.
  vq_argmax_gather<<<MROWS / 16 / WAVES_PB, 128, 0, stream>>>(
      input, weight, whi, wlo, outq, outidx);
}
